// AdapLSNet_41970420417846
// MI455X (gfx1250) — matrix operand order repaired and checked by reading, not compiled, not run on hardware
//
#include <hip/hip_runtime.h>
#include <hip/hip_bf16.h>

typedef __attribute__((ext_vector_type(16))) _Float16 v16h;
typedef __attribute__((ext_vector_type(8)))  _Float16 v8h;
typedef __attribute__((ext_vector_type(4)))  _Float16 v4h;
typedef __attribute__((ext_vector_type(8)))  float    v8f;
typedef __attribute__((ext_vector_type(4)))  int      v4i;

#define BM 128
#define BN 256
#define BK 32

#define AS1 __attribute__((address_space(1)))
#define AS3 __attribute__((address_space(3)))

#if defined(__has_builtin)
#if __has_builtin(__builtin_amdgcn_global_load_async_to_lds_b128)
#define HAVE_ASYNC_LDS 1
#endif
#endif

// 16 bytes (8 halves) per lane, global -> LDS.
// CDNA5 path: global_load_async_to_lds_b128 (ASYNCcnt, in-order completion).
__device__ __forceinline__ void copy16_g2l(const _Float16* gsrc, _Float16* ldst) {
#if defined(HAVE_ASYNC_LDS)
    __builtin_amdgcn_global_load_async_to_lds_b128(
        (AS1 v4i*)gsrc, (AS3 v4i*)ldst, 0, 0);
#else
    *(v8h*)ldst = *(const v8h*)gsrc;
#endif
}

template<int N>
__device__ __forceinline__ void async_wait() {
#if defined(HAVE_ASYNC_LDS)
#if __has_builtin(__builtin_amdgcn_s_wait_asynccnt)
    __builtin_amdgcn_s_wait_asynccnt(N);
#else
    asm volatile("s_wait_asynccnt %0" :: "i"(N) : "memory");
#endif
#endif
}

// Stage a [ROWS x 32 halves] f16 tile into LDS; ROWS*4 16B chunks spread
// over 256 threads. Works for A ([M][K] rows) and transposed weights Wt
// ([N][K] rows). ROWS=128 -> 2 async loads/thread, ROWS=256 -> 4.
template<int ROWS>
__device__ __forceinline__ void stage_tile(
    const _Float16* __restrict__ g, size_t ld, int r0, int k0,
    _Float16* lds, int tid)
{
#pragma unroll
    for (int h = 0; h < ROWS / 64; ++h) {
        int c    = h * 256 + tid;
        int row  = c >> 2;
        int part = (c & 3) * 8;
        copy16_g2l(g + (size_t)(r0 + row) * ld + k0 + part,
                   lds + row * BK + part);
    }
}

// H = elu(A @ Wt^T + bias). A: [M,K] f16, Wt: [N,K] f16 (pre-transposed),
// H: [M,N] f16. Block = 128x256 over 8 waves (2 down M x 4 across N); each
// wave owns 64x64 = 4x4 WMMA tiles (128 f32 acc VGPRs). Triple-buffered LDS
// with ASYNCcnt<=6 partial waits -> 2 K-tiles of prefetch in flight.
__global__ __launch_bounds__(256) void mlp_gemm_elu(
    const _Float16* __restrict__ A,
    const _Float16* __restrict__ Wt,
    const float*    __restrict__ bias,
    _Float16*       __restrict__ Hout,
    int M, int N, int K)
{
    __shared__ __align__(32) _Float16 As[3][BM * BK];   // 3 x 8 KB
    __shared__ __align__(32) _Float16 Bs[3][BN * BK];   // 3 x 16 KB

    const int tid  = threadIdx.x;
    const int lane = tid & 31;
    const int wid  = tid >> 5;
    const int l16  = lane & 15;
    const int lhi  = lane >> 4;
    const int waveM = (wid & 1) * 64;
    const int waveN = (wid >> 1) * 64;
    const int m0 = blockIdx.y * BM;
    const int n0 = blockIdx.x * BN;

    v8f acc[4][4] = {};

    // One K-step of WMMAs on LDS buffer `b`:
    // B 32x16 fragments: lane n holds column n, K 0-15 (lanes 0-15) or
    // K 16-31 (lanes 16-31) — contiguous 32B in Bs[n][k].
    // A 16x32 fragments: lanes 0-15 hold K 0-15, lanes 16-31 hold
    // K 16-31 of the same rows.
    auto compute_tile = [&](const _Float16* Asc, const _Float16* Bsc) {
        v16h bf[4];
#pragma unroll
        for (int j = 0; j < 4; ++j)
            bf[j] = *(const v16h*)(&Bsc[(waveN + j * 16 + l16) * BK + lhi * 16]);
#pragma unroll
        for (int i = 0; i < 4; ++i) {
            const _Float16* pa = &Asc[(waveM + i * 16 + l16) * BK + lhi * 16];
            v8h lo = *(const v8h*)(pa);
            v8h hi = *(const v8h*)(pa + 8);
            v16h af = __builtin_shufflevector(lo, hi,
                          0,1,2,3,4,5,6,7,8,9,10,11,12,13,14,15);
#pragma unroll
            for (int j = 0; j < 4; ++j)
                acc[i][j] = __builtin_amdgcn_wmma_f32_16x16x32_f16(
                    false, af, false, bf[j],
                    (short)0, acc[i][j], false, false);
        }
    };

    const int nk = K / BK;
    // prologue: stage K-tiles 0 and 1
    stage_tile<BM>(A,  K, m0, 0, As[0], tid);
    stage_tile<BN>(Wt, K, n0, 0, Bs[0], tid);
    if (nk > 1) {
        stage_tile<BM>(A,  K, m0, BK, As[1], tid);
        stage_tile<BN>(Wt, K, n0, BK, Bs[1], tid);
    }

    // steady state: unconditional asynccnt<=6 (one staged tile in flight)
    int cur = 0, nxt2 = 2;
    for (int kt = 0; kt < nk - 1; ++kt) {
        async_wait<6>();
        __syncthreads();
        if (kt + 2 < nk) {
            const int k0 = (kt + 2) * BK;
            stage_tile<BM>(A,  K, m0, k0, As[nxt2], tid);
            stage_tile<BN>(Wt, K, n0, k0, Bs[nxt2], tid);
        }
        compute_tile(As[cur], Bs[cur]);
        cur  = (cur  == 2) ? 0 : cur  + 1;
        nxt2 = (nxt2 == 2) ? 0 : nxt2 + 1;
    }
    // peeled last K-step: drain all async loads
    async_wait<0>();
    __syncthreads();
    compute_tile(As[cur], Bs[cur]);

    // epilogue: bias + ELU, store f16.
    // C/D layout: VGPR r, lanes 0-15 -> (M=r, N=lane); lanes 16-31 -> M=r+8.
    const int row0 = m0 + waveM;
    const int col0 = n0 + waveN;
#pragma unroll
    for (int j = 0; j < 4; ++j) {
        int col = col0 + j * 16 + l16;
        float bv = bias[col];
#pragma unroll
        for (int i = 0; i < 4; ++i) {
#pragma unroll
            for (int r = 0; r < 8; ++r) {
                int row = row0 + i * 16 + lhi * 8 + r;
                float v = acc[i][j][r] + bv;
                v = (v > 0.0f) ? v : (__expf(v) - 1.0f);
                Hout[(size_t)row * N + col] = (_Float16)v;
            }
        }
    }
}

// Elementwise f32 -> f16 (for x).
__global__ __launch_bounds__(256) void cvt_f32_f16(
    const float* __restrict__ in, _Float16* __restrict__ out)
{
    size_t i = ((size_t)blockIdx.x * 256 + threadIdx.x) * 4;
    float4 v = *(const float4*)(in + i);
    v4h h = { (_Float16)v.x, (_Float16)v.y, (_Float16)v.z, (_Float16)v.w };
    *(v4h*)(out + i) = h;
}

// W [K][N] f32 -> Wt [N][K] f16, LDS-tiled transpose (coalesced both sides).
__global__ __launch_bounds__(256) void w_to_f16t(
    const float* __restrict__ W, _Float16* __restrict__ Wt, int K, int N)
{
    __shared__ float tile[32][33];
    const int kb = blockIdx.y * 32, nb = blockIdx.x * 32;
    const int tx = threadIdx.x & 31, ty = threadIdx.x >> 5;   // ty: 0..7
#pragma unroll
    for (int r = 0; r < 32; r += 8)
        tile[ty + r][tx] = W[(size_t)(kb + ty + r) * N + nb + tx];
    __syncthreads();
#pragma unroll
    for (int r = 0; r < 32; r += 8)
        Wt[(size_t)(nb + ty + r) * K + kb + tx] = (_Float16)tile[tx][ty + r];
}

// out = sigmoid(H @ W2 + b2); alpha = piecewise(out). One wave per row.
__global__ __launch_bounds__(256) void mlp_head(
    const _Float16* __restrict__ H,
    const float*    __restrict__ W2,
    const float*    __restrict__ b2,
    float* __restrict__ out, float* __restrict__ alpha, int K)
{
    const int wid  = threadIdx.x >> 5;
    const int lane = threadIdx.x & 31;
    const int row  = blockIdx.x * 8 + wid;

    const _Float16* hrow = H + (size_t)row * K;
    float s = 0.0f;
    for (int j = lane; j < K; j += 32)
        s += (float)hrow[j] * W2[j];
#pragma unroll
    for (int off = 16; off > 0; off >>= 1)
        s += __shfl_xor(s, off, 32);

    if (lane == 0) {
        float z = s + b2[0];
        float o = 1.0f / (1.0f + __expf(-z));
        out[row] = o;
        const float a = 0.1f, b = 0.2f, c = 0.8f;
        float al;
        if (o <= b)      al = -a * o / b + a;
        else if (o >= c) al = a * o / (1.0f - c) + a * c / (c - 1.0f);
        else             al = 0.0f;
        alpha[row] = al;
    }
}

extern "C" void kernel_launch(void* const* d_in, const int* in_sizes, int n_in,
                              void* d_out, int out_size, void* d_ws, size_t ws_size,
                              hipStream_t stream) {
    (void)in_sizes; (void)n_in; (void)out_size; (void)ws_size;
    const float* x  = (const float*)d_in[0];   // [32768,1024]
    const float* W0 = (const float*)d_in[1];   // [1024,2048]
    const float* b0 = (const float*)d_in[2];
    const float* W1 = (const float*)d_in[3];   // [2048,2048]
    const float* b1 = (const float*)d_in[4];
    const float* W2 = (const float*)d_in[5];   // [2048,1]
    const float* b2 = (const float*)d_in[6];

    const int M = 32768, N = 2048, K0 = 1024, K1 = 2048;

    // workspace layout (f16 elements)
    _Float16* H1  = (_Float16*)d_ws;                 // 128 MB
    _Float16* H2  = H1  + (size_t)M * N;             // 128 MB
    _Float16* x16 = H2  + (size_t)M * N;             // 64 MB
    _Float16* W0t = x16 + (size_t)M * K0;            // 4 MB   [N][K0]
    _Float16* W1t = W0t + (size_t)N * K0;            // 8 MB   [N][K1]

    float* out   = (float*)d_out;
    float* alpha = out + M;

    // one-shot precision/layout conversion (memory-bound, ~10 us total)
    cvt_f32_f16<<<(size_t)M * K0 / (256 * 4), 256, 0, stream>>>(x, x16);
    w_to_f16t<<<dim3(N / 32, K0 / 32), 256, 0, stream>>>(W0, W0t, K0, N);
    w_to_f16t<<<dim3(N / 32, K1 / 32), 256, 0, stream>>>(W1, W1t, K1, N);

    dim3 grid(N / BN, M / BM);   // (8, 256)
    mlp_gemm_elu<<<grid, 256, 0, stream>>>(x16, W0t, b0, H1, M, N, K0);
    mlp_gemm_elu<<<grid, 256, 0, stream>>>(H1,  W1t, b1, H2, M, N, K1);
    mlp_head<<<M / 8, 256, 0, stream>>>(H2, W2, b2, out, alpha, K1);
}
